// SelfAttention_84817014162138
// MI455X (gfx1250) — compile-verified, matrix operand order checked by reading, not run on hardware
//
#include <hip/hip_runtime.h>
#include <hip/hip_bf16.h>

// ---------------------------------------------------------------------------
// Fused self-attention block for MI455X (gfx1250, wave32, WMMA bf16).
//   B=16, H=W=64, C=256, heads=8, hd=32.  Rows = B*H*W = 65536.
//   GEMM1 (rowsx256 @ 256x768) -> per-position 8-head softmax attention
//   -> GEMM2 (rowsx256 @ 256x256) + bias + residual.
// One workgroup = 16 rows (16 independent positions), fully fused via LDS.
// A-tile staged ONCE as bf16 (vectorized); WMMA k-loops are barrier-free;
// attention phase uses float4 LDS loads throughout.
// ---------------------------------------------------------------------------

typedef __bf16 bf16;
typedef bf16  bf16x4  __attribute__((ext_vector_type(4)));
typedef bf16  bf16x8  __attribute__((ext_vector_type(8)));
typedef bf16  bf16x16 __attribute__((ext_vector_type(16)));
typedef float floatx4 __attribute__((ext_vector_type(4)));
typedef float floatx8 __attribute__((ext_vector_type(8)));

#define CCH   256              // channels
#define NQKV  768              // 3*C
#define QKV_STRIDE 776        // f32 row stride in LDS (bank-rotated, 3104B, 16B-aligned)
#define AS    264             // bf16 row stride for A/attn-out tile (528B, 16B-aligned)

// Build a 16-element bf16 A/B fragment from two contiguous 8-element groups
// (second group lives at +16 elements; the interleaved 8..15 belong to the
// other half-wave, per the CDNA5 16-bit operand layout).
__device__ __forceinline__ bf16x16 mkfrag(const bf16* p) {
    bf16x8 lo = *(const bf16x8*)p;
    bf16x8 hi = *(const bf16x8*)(p + 16);
    return __builtin_shufflevector(lo, hi, 0,1,2,3,4,5,6,7,8,9,10,11,12,13,14,15);
}

// Transpose + convert both weight matrices to bf16:
//   wqkvT[n][k] = w_qkv[k][n]   (768 x 256)
//   wprojT[n][k] = w_proj[k][n] (256 x 256)
__global__ void prep_weights_kernel(const float* __restrict__ wqkv,
                                    const float* __restrict__ wproj,
                                    bf16* __restrict__ wqkvT,
                                    bf16* __restrict__ wprojT) {
    int idx = blockIdx.x * blockDim.x + threadIdx.x;
    const int n1 = NQKV * CCH;       // 196608
    const int n2 = CCH * CCH;        // 65536
    if (idx < n1) {
        int n = idx >> 8, k = idx & 255;
        wqkvT[idx] = (bf16)wqkv[k * NQKV + n];
    } else if (idx < n1 + n2) {
        int j = idx - n1;
        int n = j >> 8, k = j & 255;
        wprojT[j] = (bf16)wproj[k * CCH + n];
    }
}

__global__ __launch_bounds__(256) void fused_attn_kernel(
    const float* __restrict__ x,
    const bf16*  __restrict__ wqkvT,
    const float* __restrict__ bqkv,
    const bf16*  __restrict__ wprojT,
    const float* __restrict__ bproj,
    float*       __restrict__ out) {

    __shared__ float qkv[16 * QKV_STRIDE];   // 16 x 768 f32 (+pad)  ~48.5 KB
    // Dual-purpose 16x256 bf16 tile (+pad): phase 1 = bf16(x) A-tile,
    // phases 2/3 = attention output A-tile.  Lifetimes are split by the
    // barrier that follows the qkv spill, so aliasing is safe.
    __shared__ bf16  abuf[16 * AS];          // ~8.25 KB

    const int tid   = threadIdx.x;
    const int wave  = tid >> 5;
    const int lane  = tid & 31;
    const int lm    = lane & 15;        // M (A) or N (B, C/D) index within tile
    const int khalf = lane >> 4;        // which 8-wide K group this lane holds
    const int k0    = khalf * 8;
    const long rowBase = (long)blockIdx.x * 16;

    // ------------- Stage full 16x256 x-tile as bf16 (one shot) --------------
    {
        int r = tid >> 4, cg = tid & 15;             // 16 elements per thread
        const float* xp = &x[(rowBase + r) * CCH + cg * 16];
        floatx4 f0 = ((const floatx4*)xp)[0];
        floatx4 f1 = ((const floatx4*)xp)[1];
        floatx4 f2 = ((const floatx4*)xp)[2];
        floatx4 f3 = ((const floatx4*)xp)[3];
        bf16x8 p0, p1;
        #pragma unroll
        for (int i = 0; i < 4; ++i) {
            p0[i]     = (bf16)f0[i];
            p0[i + 4] = (bf16)f1[i];
            p1[i]     = (bf16)f2[i];
            p1[i + 4] = (bf16)f3[i];
        }
        bf16* dp = &abuf[r * AS + cg * 16];
        *(bf16x8*)dp       = p0;
        *(bf16x8*)(dp + 8) = p1;
    }
    __syncthreads();

    // ---------------- Phase 1: QKV projection (16 x 768, K=256) -------------
    floatx8 acc[6] = {};
    for (int kk = 0; kk < CCH; kk += 32) {           // barrier-free k-loop
        bf16x16 afrag = mkfrag(&abuf[lm * AS + kk + k0]);
        #pragma unroll
        for (int i = 0; i < 6; ++i) {
            int n = (wave * 6 + i) * 16 + lm;
            bf16x16 bfrag = mkfrag(&wqkvT[(long)n * CCH + kk + k0]);
            acc[i] = __builtin_amdgcn_wmma_f32_16x16x32_bf16(
                false, afrag, false, bfrag, (short)0, acc[i], false, false);
        }
    }
    // spill qkv tile (+bias) to LDS.  D layout: VGPR r -> M = r + 8*khalf, N = lm.
    #pragma unroll
    for (int i = 0; i < 6; ++i) {
        int n = (wave * 6 + i) * 16 + lm;
        float bias = bqkv[n];
        #pragma unroll
        for (int r = 0; r < 8; ++r)
            qkv[(r + khalf * 8) * QKV_STRIDE + n] = acc[i][r] + bias;
    }
    // warm the residual x lines for the epilogue while we sync
    __builtin_prefetch(&x[(rowBase + lm) * CCH + wave * 32 + khalf * 16], 0, 1);
    __syncthreads();   // qkv visible AND all phase-1 abuf reads complete

    // ---------------- Phase 2: per-position 8-head attention (f32) ----------
    if (tid < 128) {
        int s = tid >> 3;                     // position within tile
        int h = tid & 7;                      // head (query row)
        const float* base = &qkv[s * QKV_STRIDE];

        floatx4 qv[8];                        // q row, 8 x float4
        #pragma unroll
        for (int j = 0; j < 8; ++j)
            qv[j] = *(const floatx4*)(base + h * 32 + j * 4);

        float logits[8];
        const float scale = 0.17677669529663687f;   // 32^-0.5
        #pragma unroll
        for (int m = 0; m < 8; ++m) {
            const floatx4* kr = (const floatx4*)(base + CCH + m * 32);
            floatx4 a4 = {};
            #pragma unroll
            for (int j = 0; j < 8; ++j)
                a4 = a4 + qv[j] * kr[j];
            logits[m] = (a4[0] + a4[1] + a4[2] + a4[3]) * scale;
        }
        float mx = logits[0];
        #pragma unroll
        for (int m = 1; m < 8; ++m) mx = fmaxf(mx, logits[m]);
        float sum = 0.f;
        #pragma unroll
        for (int m = 0; m < 8; ++m) { logits[m] = __expf(logits[m] - mx); sum += logits[m]; }
        float inv = 1.f / sum;

        #pragma unroll
        for (int j = 0; j < 8; ++j) {         // out row, 8 x float4
            floatx4 o = {};
            #pragma unroll
            for (int m = 0; m < 8; ++m) {
                floatx4 vv = *(const floatx4*)(base + 2 * CCH + m * 32 + j * 4);
                o = o + logits[m] * vv;
            }
            o = o * inv;
            bf16x4 ob;
            #pragma unroll
            for (int c = 0; c < 4; ++c) ob[c] = (bf16)o[c];
            *(bf16x4*)&abuf[s * AS + h * 32 + j * 4] = ob;
        }
    }
    __syncthreads();

    // ---------------- Phase 3: output projection + residual -----------------
    floatx8 acc2[2] = {};
    for (int kk = 0; kk < CCH; kk += 32) {           // barrier-free k-loop
        bf16x16 afrag = mkfrag(&abuf[lm * AS + kk + k0]);
        #pragma unroll
        for (int i = 0; i < 2; ++i) {
            int n = (wave * 2 + i) * 16 + lm;
            bf16x16 bfrag = mkfrag(&wprojT[(long)n * CCH + kk + k0]);
            acc2[i] = __builtin_amdgcn_wmma_f32_16x16x32_bf16(
                false, afrag, false, bfrag, (short)0, acc2[i], false, false);
        }
    }
    #pragma unroll
    for (int i = 0; i < 2; ++i) {
        int n = (wave * 2 + i) * 16 + lm;
        float bias = bproj[n];
        #pragma unroll
        for (int r = 0; r < 8; ++r) {
            long idx = (rowBase + r + khalf * 8) * CCH + n;
            out[idx] = x[idx] + acc2[i][r] + bias;
        }
    }
}

extern "C" void kernel_launch(void* const* d_in, const int* in_sizes, int n_in,
                              void* d_out, int out_size, void* d_ws, size_t ws_size,
                              hipStream_t stream) {
    const float* x     = (const float*)d_in[0];
    const float* wqkv  = (const float*)d_in[1];
    const float* bqkv  = (const float*)d_in[2];
    const float* wproj = (const float*)d_in[3];
    const float* bproj = (const float*)d_in[4];
    float* out = (float*)d_out;

    bf16* wqkvT  = (bf16*)d_ws;                  // 768*256 bf16 = 384 KB
    bf16* wprojT = wqkvT + NQKV * CCH;           // 256*256 bf16 = 128 KB

    const int prepN = NQKV * CCH + CCH * CCH;    // 262144
    prep_weights_kernel<<<(prepN + 255) / 256, 256, 0, stream>>>(wqkv, wproj, wqkvT, wprojT);

    // 65536 rows / 16 rows per block = 4096 blocks, 8 waves each
    fused_attn_kernel<<<4096, 256, 0, stream>>>(x, wqkvT, bqkv, wprojT, bproj, out);
}
